// DecoderLayer_82334523065080
// MI455X (gfx1250) — matrix operand order repaired and checked by reading, not compiled, not run on hardware
//
#include <hip/hip_runtime.h>
#include <hip/hip_bf16.h>
#include <stdint.h>
#include <stddef.h>

// ---------------------------------------------------------------------------
// MI455X (gfx1250) transformer decoder layer.
// All matmul work mapped to v_wmma_f32_16x16x32_bf16 (wave32 WMMA, f32 accum).
// B tile staging uses gfx1250 async global->LDS loads when available.
// ---------------------------------------------------------------------------

typedef __attribute__((ext_vector_type(8)))  float  f32x8;
typedef __attribute__((ext_vector_type(8)))  __bf16 bf16x8;
typedef __attribute__((ext_vector_type(16))) __bf16 bf16x16;
typedef int v4i_ __attribute__((vector_size(16)));   // matches builtin's V4i

union FragU { bf16x16 v; bf16x8 h[2]; };

#define NEG_INF (-__builtin_inff())

#if __has_builtin(__builtin_amdgcn_global_load_async_to_lds_b128)
#define HAVE_ASYNC_LDS 1
#else
#define HAVE_ASYNC_LDS 0
#endif

// Native converts: backend emits v_cvt_*bf16_f32 (RNE) instead of SW bit ops.
static __device__ __forceinline__ __bf16 f2bf(float f) { return (__bf16)f; }

static __device__ __forceinline__ f32x8 wmma_bf16(bf16x16 a, bf16x16 b, f32x8 c) {
    return __builtin_amdgcn_wmma_f32_16x16x32_bf16(false, a, false, b,
                                                   (short)0, c, false, false);
}

static __device__ __forceinline__ void async_g2l_b128(const void* g, void* l) {
#if HAVE_ASYNC_LDS
    // Signature (probe-confirmed via diagnostic): (v4i AS1*, v4i AS3*, Ii, Ii).
    // Low 32 bits of a generic __shared__ address are the LDS byte offset.
    __builtin_amdgcn_global_load_async_to_lds_b128(
        (__attribute__((address_space(1))) v4i_*)(uintptr_t)g,
        (__attribute__((address_space(3))) v4i_*)(unsigned)(uintptr_t)l,
        0, 0);
#else
    (void)g; (void)l;
#endif
}

static __device__ __forceinline__ void wait_async0() {
#if HAVE_ASYNC_LDS
#if __has_builtin(__builtin_amdgcn_s_wait_asynccnt)
    __builtin_amdgcn_s_wait_asynccnt(0);
#else
    asm volatile("s_wait_asynccnt 0" ::: "memory");
#endif
#endif
}

// ---------------------------------------------------------------------------
// Weight transpose + convert:  W[K][N] f32  ->  Wt[N][K] bf16
// ---------------------------------------------------------------------------
__global__ __launch_bounds__(256) void k_wT(const float* __restrict__ W,
                                            __bf16* __restrict__ Wt,
                                            int K, int N) {
    size_t idx = (size_t)blockIdx.x * 256 + threadIdx.x;  // coalesced read
    int k = (int)(idx / (size_t)N);
    int n = (int)(idx % (size_t)N);
    Wt[(size_t)n * K + k] = f2bf(W[idx]);
}

// ---------------------------------------------------------------------------
// GEMM:  C[M][N] = A[M][K] (f32) * Wt^T  + bias,  optional ReLU.
// Wt is [N][K] bf16.  BM=BN=128, BK=32, 8 waves, wave tile 64x32 (8 WMMA/step).
// Double-buffered LDS; B tiles staged with async global->LDS when available.
// ---------------------------------------------------------------------------
#define GBM 128
#define GBN 128
#define GBK 32
#define GAK 40   // LDS row stride (elements): 80B, 16B-aligned, 20-bank stride

__global__ __launch_bounds__(256) void k_gemm(const float*  __restrict__ A,
                                              const __bf16* __restrict__ Bt,
                                              const float*  __restrict__ bias,
                                              float*        __restrict__ C,
                                              int M, int N, int K, int relu) {
    __shared__ __bf16 Al[2][GBM * GAK];
    __shared__ __bf16 Bl[2][GBN * GAK];

    const int t    = threadIdx.x;
    const int lane = t & 31;
    const int wid  = t >> 5;
    const int wm   = wid >> 2;          // 0..1 (M direction)
    const int wn   = wid & 3;           // 0..3 (N direction)
    const int hi16 = (lane >> 4) & 1;   // lane half (ISA fragment layouts)
    const int l16  = lane & 15;

    const int rowBase = blockIdx.y * GBM;
    const int colBase = blockIdx.x * GBN;

    // load next A tile into registers (no LDS store -> no forced waits)
    auto loadA = [&](int kt, float4 ar[4]) {
        const float* Ag = A + (size_t)rowBase * K + (size_t)kt * GBK;
#pragma unroll
        for (int i = 0; i < 4; ++i) {
            int c  = t + i * 256;        // 1024 float4 chunks (128 rows x 8)
            int r  = c >> 3;
            int c4 = c & 7;
            ar[i] = *(const float4*)(Ag + (size_t)r * K + c4 * 4);
        }
    };
    auto storeA = [&](__bf16* dst, const float4 ar[4]) {
#pragma unroll
        for (int i = 0; i < 4; ++i) {
            int c  = t + i * 256;
            int r  = c >> 3;
            int c4 = c & 7;
            __bf16* d = &dst[r * GAK + c4 * 4];
            d[0] = f2bf(ar[i].x); d[1] = f2bf(ar[i].y);
            d[2] = f2bf(ar[i].z); d[3] = f2bf(ar[i].w);
        }
    };
#if HAVE_ASYNC_LDS
    auto stageB = [&](int kt, __bf16* dst) {   // async global->LDS, 16B/lane
        const __bf16* Bg = Bt + (size_t)colBase * K + (size_t)kt * GBK;
#pragma unroll
        for (int i = 0; i < 2; ++i) {
            int c  = t + i * 256;        // 512 bf16x8 chunks (128 rows x 4)
            int n  = c >> 2;
            int c8 = c & 3;
            async_g2l_b128(Bg + (size_t)n * K + c8 * 8, &dst[n * GAK + c8 * 8]);
        }
    };
#else
    auto loadB = [&](int kt, bf16x8 br[2]) {
        const __bf16* Bg = Bt + (size_t)colBase * K + (size_t)kt * GBK;
#pragma unroll
        for (int i = 0; i < 2; ++i) {
            int c  = t + i * 256;
            int n  = c >> 2;
            int c8 = c & 3;
            br[i] = *(const bf16x8*)(Bg + (size_t)n * K + c8 * 8);
        }
    };
    auto storeB = [&](__bf16* dst, const bf16x8 br[2]) {
#pragma unroll
        for (int i = 0; i < 2; ++i) {
            int c  = t + i * 256;
            int n  = c >> 2;
            int c8 = c & 3;
            *(bf16x8*)(&dst[n * GAK + c8 * 8]) = br[i];
        }
    };
#endif

    f32x8 acc[4][2];
#pragma unroll
    for (int mi = 0; mi < 4; ++mi)
#pragma unroll
        for (int ni = 0; ni < 2; ++ni)
#pragma unroll
            for (int j = 0; j < 8; ++j) acc[mi][ni][j] = 0.0f;

    const int ktiles = K / GBK;

    // ---- prologue: stage tile 0
    {
        float4 ar[4];
        loadA(0, ar);
#if HAVE_ASYNC_LDS
        stageB(0, Bl[0]);
        storeA(Al[0], ar);
#else
        bf16x8 br[2];
        loadB(0, br);
        storeA(Al[0], ar);
        storeB(Bl[0], br);
#endif
        wait_async0();
        __syncthreads();
    }

    for (int kt = 0; kt < ktiles; ++kt) {
        const int cur  = kt & 1;
        const int nxt  = cur ^ 1;
        const bool more = (kt + 1) < ktiles;

        float4 ar[4];
#if !HAVE_ASYNC_LDS
        bf16x8 br[2];
#endif
        if (more) {                       // overlap next-tile loads with WMMA
            loadA(kt + 1, ar);
#if HAVE_ASYNC_LDS
            stageB(kt + 1, Bl[nxt]);
#else
            loadB(kt + 1, br);
#endif
        }
        if (kt + 2 < ktiles)              // global_prefetch_b8 of tile kt+2
            __builtin_prefetch(A + (size_t)(rowBase + (t >> 1)) * K +
                                   (size_t)(kt + 2) * GBK + (t & 1) * 16, 0, 1);

        // ---- fragments (ISA 16-bit A/B layouts) from buffer `cur`
        const __bf16* Ac = Al[cur];
        const __bf16* Bc = Bl[cur];
        bf16x16 af[4], bfr[2];
        const int alo = hi16 ? 8 : 0;
#pragma unroll
        for (int mi = 0; mi < 4; ++mi) {
            int r = wm * 64 + mi * 16 + l16;
            FragU u;
            u.h[0] = *(const bf16x8*)&Ac[r * GAK + alo];
            u.h[1] = *(const bf16x8*)&Ac[r * GAK + alo + 16];
            af[mi] = u.v;
        }
        const int bko = hi16 ? 8 : 0;
#pragma unroll
        for (int ni = 0; ni < 2; ++ni) {
            int n = wn * 32 + ni * 16 + l16;
            FragU u;
            u.h[0] = *(const bf16x8*)&Bc[n * GAK + bko];
            u.h[1] = *(const bf16x8*)&Bc[n * GAK + bko + 16];
            bfr[ni] = u.v;
        }
#pragma unroll
        for (int mi = 0; mi < 4; ++mi)
#pragma unroll
            for (int ni = 0; ni < 2; ++ni)
                acc[mi][ni] = wmma_bf16(af[mi], bfr[ni], acc[mi][ni]);

        if (more) {
            storeA(Al[nxt], ar);
#if !HAVE_ASYNC_LDS
            storeB(Bl[nxt], br);
#endif
            wait_async0();
        }
        __syncthreads();
    }

    // ---- epilogue: bias (+ReLU), f32 store
#pragma unroll
    for (int mi = 0; mi < 4; ++mi)
#pragma unroll
        for (int ni = 0; ni < 2; ++ni) {
            int col = colBase + wn * 32 + ni * 16 + l16;
            float bb = bias[col];
#pragma unroll
            for (int r = 0; r < 8; ++r) {
                int row = rowBase + wm * 64 + mi * 16 + r + hi16 * 8;
                float v = acc[mi][ni][r] + bb;
                if (relu) v = fmaxf(v, 0.0f);
                C[(size_t)row * N + col] = v;
            }
        }
}

// ---------------------------------------------------------------------------
// Flash attention.  Q,K,V,Out are [B,S,D] f32; head h occupies cols h*64..+63.
// Block = 8 waves; wave owns a 16-row Q tile (block covers 128 q rows of one
// (b,h)).  32-key chunks: QK^T = 4 WMMAs, PV = 4 WMMAs, online softmax.
// ---------------------------------------------------------------------------
#define FD   1024
#define FH   16
#define FDH  64

__global__ __launch_bounds__(256) void k_flash(const float* __restrict__ Q,
                                               const float* __restrict__ Kv,
                                               const float* __restrict__ V,
                                               float*       __restrict__ Out,
                                               int Sq, int Skv, int causal) {
    __shared__ __bf16 Ql[128 * 72];      // Q tile, stride 72 (144B rows)
    __shared__ __bf16 Kl[32 * 72];       // K chunk [key][dh]
    __shared__ __bf16 VT[64 * 40];       // V chunk transposed [dh][key]
    __shared__ __bf16 Pl[8 * 16 * 40];   // per-wave P staging (C->A relayout)

    const int t    = threadIdx.x;
    const int lane = t & 31;
    const int wid  = t >> 5;
    const int hi16 = (lane >> 4) & 1;
    const int l16  = lane & 15;

    const int bh = blockIdx.y;
    const int b  = bh >> 4;
    const int h  = bh & 15;
    const size_t hoff = (size_t)h * FDH;
    const int qBlock = blockIdx.x * 128;
    const int qrow0  = wid * 16;

    // ---- stage Q (scaled by 1/sqrt(dh) = 1/8), convert to bf16
    const float qs = 0.125f;
    {
        float4 qr[8];
#pragma unroll
        for (int i = 0; i < 8; ++i) {
            int c  = t + i * 256;        // 2048 float4 chunks: 128 rows x 16
            int r  = c >> 4;
            int c4 = c & 15;
            qr[i] = *(const float4*)(Q + ((size_t)(b * Sq + qBlock + r)) * FD
                                       + hoff + c4 * 4);
        }
#pragma unroll
        for (int i = 0; i < 8; ++i) {
            int c  = t + i * 256;
            int r  = c >> 4;
            int c4 = c & 15;
            __bf16* dst = &Ql[r * 72 + c4 * 4];
            dst[0] = f2bf(qr[i].x * qs); dst[1] = f2bf(qr[i].y * qs);
            dst[2] = f2bf(qr[i].z * qs); dst[3] = f2bf(qr[i].w * qs);
        }
    }
    __syncthreads();

    // ---- Q fragments (kept in VGPRs for the whole kernel)
    bf16x16 qf[2];
    {
        const int lo = hi16 ? 8 : 0;
        const int r  = qrow0 + l16;
#pragma unroll
        for (int kk = 0; kk < 2; ++kk) {
            FragU u;
            u.h[0] = *(const bf16x8*)&Ql[r * 72 + kk * 32 + lo];
            u.h[1] = *(const bf16x8*)&Ql[r * 72 + kk * 32 + lo + 16];
            qf[kk] = u.v;
        }
    }

    float m[8], l[8];
    f32x8 Oacc[4];
#pragma unroll
    for (int r = 0; r < 8; ++r) { m[r] = NEG_INF; l[r] = 0.0f; }
#pragma unroll
    for (int d = 0; d < 4; ++d)
#pragma unroll
        for (int j = 0; j < 8; ++j) Oacc[d][j] = 0.0f;

    __bf16* Pw = &Pl[wid * 16 * 40];
    const int nch = causal ? (blockIdx.x + 1) * 4 : (Skv / 32);

    for (int ch = 0; ch < nch; ++ch) {
        const int kb = ch * 32;
        __syncthreads();
        // ---- stage K chunk [32][64] and V^T [64][32]; batch loads first
        {
            float4 kr[2], vr[2];
#pragma unroll
            for (int i = 0; i < 2; ++i) {
                int c  = t + i * 256;    // 512 float4 chunks
                int r  = c >> 4;
                int c4 = c & 15;
                kr[i] = *(const float4*)(Kv + ((size_t)(b * Skv + kb + r)) * FD
                                            + hoff + c4 * 4);
                vr[i] = *(const float4*)(V  + ((size_t)(b * Skv + kb + r)) * FD
                                            + hoff + c4 * 4);
            }
#pragma unroll
            for (int i = 0; i < 2; ++i) {
                int c  = t + i * 256;
                int r  = c >> 4;
                int c4 = c & 15;
                __bf16* dst = &Kl[r * 72 + c4 * 4];
                dst[0] = f2bf(kr[i].x); dst[1] = f2bf(kr[i].y);
                dst[2] = f2bf(kr[i].z); dst[3] = f2bf(kr[i].w);
                int d0 = c4 * 4;
                VT[(d0 + 0) * 40 + r] = f2bf(vr[i].x);
                VT[(d0 + 1) * 40 + r] = f2bf(vr[i].y);
                VT[(d0 + 2) * 40 + r] = f2bf(vr[i].z);
                VT[(d0 + 3) * 40 + r] = f2bf(vr[i].w);
            }
        }
        __syncthreads();

        // ---- scores: S(16x32) = Q(16x64) . K^T, two 16x16 accums
        f32x8 s0, s1;
#pragma unroll
        for (int j = 0; j < 8; ++j) { s0[j] = 0.0f; s1[j] = 0.0f; }
#pragma unroll
        for (int kk = 0; kk < 2; ++kk) {
            const int ko = kk * 32 + (hi16 ? 8 : 0);
            FragU u0, u1;
            u0.h[0] = *(const bf16x8*)&Kl[(l16)      * 72 + ko];
            u0.h[1] = *(const bf16x8*)&Kl[(l16)      * 72 + ko + 16];
            u1.h[0] = *(const bf16x8*)&Kl[(16 + l16) * 72 + ko];
            u1.h[1] = *(const bf16x8*)&Kl[(16 + l16) * 72 + ko + 16];
            s0 = wmma_bf16(qf[kk], u0.v, s0);
            s1 = wmma_bf16(qf[kk], u1.v, s1);
        }

        // ---- online softmax (rows in VGPR idx + lane half; cols across lanes)
#pragma unroll
        for (int r = 0; r < 8; ++r) {
            const int qrow = qBlock + qrow0 + r + hi16 * 8;
            float a  = s0[r];
            float bs = s1[r];
            if (causal) {
                if (kb + l16 > qrow)      a  = NEG_INF;
                if (kb + 16 + l16 > qrow) bs = NEG_INF;
            }
            float mx = fmaxf(a, bs);
#pragma unroll
            for (int off = 1; off < 16; off <<= 1)
                mx = fmaxf(mx, __shfl_xor(mx, off, 32));
            const float mnew  = fmaxf(m[r], mx);
            const float alpha = (m[r] == mnew) ? 1.0f : __expf(m[r] - mnew);
            const float p0 = (a  == NEG_INF) ? 0.0f : __expf(a  - mnew);
            const float p1 = (bs == NEG_INF) ? 0.0f : __expf(bs - mnew);
            float rs = p0 + p1;
#pragma unroll
            for (int off = 1; off < 16; off <<= 1)
                rs += __shfl_xor(rs, off, 32);
            l[r] = l[r] * alpha + rs;
            m[r] = mnew;
#pragma unroll
            for (int d = 0; d < 4; ++d) Oacc[d][r] *= alpha;
            const int prow = r + hi16 * 8;
            Pw[prow * 40 + l16]      = f2bf(p0);
            Pw[prow * 40 + 16 + l16] = f2bf(p1);
        }

        // ---- P fragment (A layout) and PV accumulate
        FragU up;
        {
            const int lo = hi16 ? 8 : 0;
            up.h[0] = *(const bf16x8*)&Pw[l16 * 40 + lo];
            up.h[1] = *(const bf16x8*)&Pw[l16 * 40 + lo + 16];
        }
        const int vo = hi16 ? 8 : 0;
#pragma unroll
        for (int d = 0; d < 4; ++d) {
            FragU uv;
            const int n = d * 16 + l16;
            uv.h[0] = *(const bf16x8*)&VT[n * 40 + vo];
            uv.h[1] = *(const bf16x8*)&VT[n * 40 + vo + 16];
            Oacc[d] = wmma_bf16(up.v, uv.v, Oacc[d]);
        }
    }

    // ---- finalize: O /= l, store f32
#pragma unroll
    for (int r = 0; r < 8; ++r) {
        const float inv = (l[r] > 0.0f) ? (1.0f / l[r]) : 0.0f;
        const int qrow = qBlock + qrow0 + r + hi16 * 8;
#pragma unroll
        for (int d = 0; d < 4; ++d)
            Out[((size_t)(b * Sq + qrow)) * FD + hoff + d * 16 + l16] =
                Oacc[d][r] * inv;
    }
}

// ---------------------------------------------------------------------------
// LayerNorm(x + res) * gamma + beta,  D = 1024, one block per row.
// ---------------------------------------------------------------------------
__global__ __launch_bounds__(256) void k_ln(const float* __restrict__ x,
                                            const float* __restrict__ res,
                                            const float* __restrict__ g,
                                            const float* __restrict__ be,
                                            float* __restrict__ out) {
    __shared__ float r0[8], r1[8];
    const size_t base = (size_t)blockIdx.x * 1024;
    const int lane = threadIdx.x & 31, wid = threadIdx.x >> 5;

    float v[4], s = 0.0f, ss = 0.0f;
#pragma unroll
    for (int i = 0; i < 4; ++i) {
        int c = threadIdx.x + i * 256;
        v[i] = x[base + c] + res[base + c];
        s += v[i]; ss += v[i] * v[i];
    }
#pragma unroll
    for (int off = 1; off < 32; off <<= 1) {
        s  += __shfl_xor(s, off, 32);
        ss += __shfl_xor(ss, off, 32);
    }
    if (lane == 0) { r0[wid] = s; r1[wid] = ss; }
    __syncthreads();
    s = 0.0f; ss = 0.0f;
#pragma unroll
    for (int w = 0; w < 8; ++w) { s += r0[w]; ss += r1[w]; }
    const float mean = s * (1.0f / 1024.0f);
    const float var  = ss * (1.0f / 1024.0f) - mean * mean;
    const float rstd = rsqrtf(var + 1e-6f);
#pragma unroll
    for (int i = 0; i < 4; ++i) {
        int c = threadIdx.x + i * 256;
        out[base + c] = (v[i] - mean) * rstd * g[c] + be[c];
    }
}

// ---------------------------------------------------------------------------
// Host-side orchestration (graph-capture safe: only kernel launches).
// ---------------------------------------------------------------------------
extern "C" void kernel_launch(void* const* d_in, const int* in_sizes, int n_in,
                              void* d_out, int out_size, void* d_ws, size_t ws_size,
                              hipStream_t stream) {
    (void)in_sizes; (void)n_in; (void)out_size; (void)ws_size;
    const int Bb = 4, S = 1024, Dm = 1024, Ff = 4096;
    const int Mrows = Bb * S;                       // 4096
    const size_t nBSD = (size_t)Bb * S * Dm;        // 4M f32
    const size_t nBSF = (size_t)Bb * S * Ff;        // 16M f32

    const float* X0  = (const float*)d_in[0];
    const float* Enc = (const float*)d_in[1];

    // workspace layout
    char* w = (char*)d_ws;
    size_t off = 0;
    auto take = [&](size_t bytes) -> void* {
        void* p = w + off;
        off = (off + bytes + 255) & ~(size_t)255;
        return p;
    };
    __bf16* wT[10];
    for (int i = 0; i < 8; ++i) wT[i] = (__bf16*)take((size_t)Dm * Dm * 2);
    wT[8] = (__bf16*)take((size_t)Dm * Ff * 2);     // w_ff1^T
    wT[9] = (__bf16*)take((size_t)Ff * Dm * 2);     // w_ff2^T
    float* bufA = (float*)take(nBSD * 4);
    float* bufB = (float*)take(nBSD * 4);
    float* bufC = (float*)take(nBSD * 4);
    float* bufD = (float*)take(nBSD * 4);
    float* bufE = (float*)take(nBSD * 4);
    float* bufF = (float*)take(nBSD * 4);
    float* bufH = (float*)take(nBSD * 4);
    float* bufG = (float*)take(nBSF * 4);

    // ---- convert weights to transposed bf16
    const int widx[10] = {4, 6, 8, 10, 12, 14, 16, 18, 20, 22};
    for (int i = 0; i < 10; ++i) {
        int K = (i == 9) ? Ff : Dm;
        int N = (i == 8) ? Ff : Dm;
        size_t total = (size_t)K * N;
        k_wT<<<dim3((unsigned)(total / 256)), 256, 0, stream>>>(
            (const float*)d_in[widx[i]], wT[i], K, N);
    }

    auto gemm = [&](const float* A, const __bf16* Bt, const float* bias,
                    float* C, int N, int K, int relu) {
        k_gemm<<<dim3(N / GBN, Mrows / GBM), 256, 0, stream>>>(
            A, Bt, bias, C, Mrows, N, K, relu);
    };
    const dim3 fgrid(S / 128, Bb * FH);

    // ---- self-attention (causal)
    gemm(X0, wT[0], (const float*)d_in[5],  bufA, Dm, Dm, 0);   // Q
    gemm(X0, wT[1], (const float*)d_in[7],  bufB, Dm, Dm, 0);   // K
    gemm(X0, wT[2], (const float*)d_in[9],  bufC, Dm, Dm, 0);   // V
    k_flash<<<fgrid, 256, 0, stream>>>(bufA, bufB, bufC, bufD, S, S, 1);
    gemm(bufD, wT[3], (const float*)d_in[11], bufE, Dm, Dm, 0); // out proj
    k_ln<<<Mrows, 256, 0, stream>>>(bufE, X0,
        (const float*)d_in[24], (const float*)d_in[25], bufF);  // x1

    // ---- cross-attention (no mask)
    gemm(bufF, wT[4], (const float*)d_in[13], bufA, Dm, Dm, 0); // Q2
    gemm(Enc,  wT[5], (const float*)d_in[15], bufB, Dm, Dm, 0); // K2
    gemm(Enc,  wT[6], (const float*)d_in[17], bufC, Dm, Dm, 0); // V2
    k_flash<<<fgrid, 256, 0, stream>>>(bufA, bufB, bufC, bufD, S, S, 0);
    gemm(bufD, wT[7], (const float*)d_in[19], bufE, Dm, Dm, 0);
    k_ln<<<Mrows, 256, 0, stream>>>(bufE, bufF,
        (const float*)d_in[26], (const float*)d_in[27], bufH);  // x2

    // ---- FFN
    gemm(bufH, wT[8], (const float*)d_in[21], bufG, Ff, Dm, 1); // relu
    gemm(bufG, wT[9], (const float*)d_in[23], bufE, Dm, Ff, 0);
    k_ln<<<Mrows, 256, 0, stream>>>(bufE, bufH,
        (const float*)d_in[28], (const float*)d_in[29], (float*)d_out);
}
